// RNNEmbedder_85744727097349
// MI455X (gfx1250) — matrix operand order repaired and checked by reading, not compiled, not run on hardware
//
#include <hip/hip_runtime.h>
#include <hip/hip_bf16.h>

// ---------------------------------------------------------------------------
// Bidirectional GRU, T=8192, Y=512, H=2048, fp32 in/out.
// Phase 1: igates GEMM via v_wmma_f32_16x16x32_bf16 with bf16 hi/lo split.
// Phase 2: persistent-grid recurrent scan, device-wide atomic barrier/step,
//          L2-resident w_hh matvec with b128 loads + wave32 shuffle reduce,
//          global_prefetch_b8 look-ahead on the weight streams.
// ---------------------------------------------------------------------------

#define T_LEN 8192
#define Y_DIM 512
#define H_DIM 2048
#define G3 (3 * H_DIM)   // 6144
#define OUTW (2 * H_DIM) // 4096

#define NB 256 // scan grid (128 blocks fwd, 128 blocks rev)
#define NT 256 // scan block (8 waves)

typedef __attribute__((ext_vector_type(16))) __bf16 v16bf;
typedef __attribute__((ext_vector_type(8))) float v8f;

__device__ __forceinline__ void split_bf16(float x, __bf16& hi, __bf16& lo) {
    hi = (__bf16)x;
    lo = (__bf16)(x - (float)hi);
}

// ---------------------------------------------------------------------------
// Phase 1: igates[dir][s][3H] = seq[t(s)] @ w_ih^T + b   (t = s fwd, T-1-s rev)
// Block = 256 thr = 8 waves arranged 4(M) x 2(N); tile 64(M) x 128(N), K=512.
// Each wave: 16x64 output via 4 N-tiles, 3 WMMAs (hi*hi, hi*lo, lo*hi) per
// 16x16x32 tile per K-chunk.
// ---------------------------------------------------------------------------
__global__ __launch_bounds__(256) void igates_gemm(
    const float* __restrict__ seq,
    const float* __restrict__ wf, const float* __restrict__ bf,
    const float* __restrict__ wr, const float* __restrict__ br,
    float* __restrict__ igf, float* __restrict__ igr)
{
    const int dir = blockIdx.z;
    const float* w    = dir ? wr : wf;
    const float* bias = dir ? br : bf;
    float* ig         = dir ? igr : igf;

    const int lane = threadIdx.x & 31;
    const int wave = threadIdx.x >> 5;
    const int wm = wave & 3;          // 0..3 over M
    const int wn = wave >> 2;         // 0..1 over N
    const int M0 = blockIdx.y * 64 + wm * 16;
    const int N0 = blockIdx.x * 128 + wn * 64;

    const int hi16 = lane >> 4;       // half-wave select
    const int l16  = lane & 15;

    const int row  = M0 + l16;                          // scan-step index
    const int srow = dir ? (T_LEN - 1 - row) : row;     // seq row
    const float* arow = seq + (size_t)srow * Y_DIM;

    const int kbA = hi16 * 8;         // A K-base within chunk for this half-wave
    const int kbB = hi16 * 8;         // B K-base within chunk

    v8f acc[4] = {};

    for (int kc = 0; kc < Y_DIM; kc += 32) {
        // ---- A fragment (16x32 bf16): lane l16 = row M, halves i -> K
        //      K(i) = (i>>3)*16 + kbA + (i&7)
        v16bf ahi, alo;
        {
            const float* p0 = arow + kc + kbA;        // K = kbA + 0..7
            const float* p1 = arow + kc + 16 + kbA;   // K = 16 + kbA + 0..7
#pragma unroll
            for (int i = 0; i < 8; ++i) {
                __bf16 h_, l_;
                split_bf16(p0[i], h_, l_); ahi[i] = h_;     alo[i] = l_;
                split_bf16(p1[i], h_, l_); ahi[8 + i] = h_; alo[8 + i] = l_;
            }
        }
#pragma unroll
        for (int tnum = 0; tnum < 4; ++tnum) {
            // ---- B fragment (32x16 bf16): lane l16 = col N, K = (i>>3)*16 + kbB + (i&7)
            //      B[k][n] = w_ih[n][k] -> contiguous along K in w row n
            const int n = N0 + tnum * 16 + l16;
            const float* bp = w + (size_t)n * Y_DIM + kc;
            v16bf bhi, blo;
#pragma unroll
            for (int i = 0; i < 16; ++i) {
                __bf16 h_, l_;
                split_bf16(bp[(i >> 3) * 16 + kbB + (i & 7)], h_, l_);
                bhi[i] = h_; blo[i] = l_;
            }
            acc[tnum] = __builtin_amdgcn_wmma_f32_16x16x32_bf16(
                false, ahi, false, bhi, (short)0, acc[tnum], false, false);
            acc[tnum] = __builtin_amdgcn_wmma_f32_16x16x32_bf16(
                false, ahi, false, blo, (short)0, acc[tnum], false, false);
            acc[tnum] = __builtin_amdgcn_wmma_f32_16x16x32_bf16(
                false, alo, false, bhi, (short)0, acc[tnum], false, false);
        }
    }

    // ---- epilogue: C/D layout lane -> N = l16, VGPR r -> M = r + 8*hi16
#pragma unroll
    for (int tnum = 0; tnum < 4; ++tnum) {
        const int n = N0 + tnum * 16 + l16;
        const float bv = bias[n];
#pragma unroll
        for (int r = 0; r < 8; ++r) {
            const int m = M0 + r + 8 * hi16;
            ig[(size_t)m * G3 + n] = acc[tnum][r] + bv;
        }
    }
}

// ---------------------------------------------------------------------------
// Init: zero h ping-pong state + barrier words (ws is poisoned by harness).
// ---------------------------------------------------------------------------
__global__ void scan_init(float* __restrict__ hbuf, unsigned* __restrict__ bar)
{
    const int i = blockIdx.x * blockDim.x + threadIdx.x;
    if (i < 4 * H_DIM) hbuf[i] = 0.0f; // [buf(2)][dir(2)][H]
    if (i < 4) bar[i] = 0u;
}

// ---------------------------------------------------------------------------
// Device-wide sense barrier (persistent grid, all NB blocks co-resident:
// 10KB LDS/block, light VGPRs -> trivially resident on any CDNA5 part).
// ---------------------------------------------------------------------------
__device__ __forceinline__ void grid_sync(unsigned* bar, unsigned target)
{
    __threadfence(); // release our h stores to device scope
    __syncthreads();
    if (threadIdx.x == 0) {
        unsigned arr = __hip_atomic_fetch_add(&bar[0], 1u, __ATOMIC_ACQ_REL,
                                              __HIP_MEMORY_SCOPE_AGENT);
        if (arr == NB - 1) {
            __hip_atomic_store(&bar[0], 0u, __ATOMIC_RELAXED,
                               __HIP_MEMORY_SCOPE_AGENT);
            __hip_atomic_store(&bar[1], target, __ATOMIC_RELEASE,
                               __HIP_MEMORY_SCOPE_AGENT);
        } else {
            while (__hip_atomic_load(&bar[1], __ATOMIC_ACQUIRE,
                                     __HIP_MEMORY_SCOPE_AGENT) < target) {
                __builtin_amdgcn_s_sleep(2);
            }
        }
    }
    __threadfence(); // acquire
    __syncthreads();
}

__device__ __forceinline__ float sigm_f(float x) {
    return 1.0f / (1.0f + __expf(-x));
}
__device__ __forceinline__ float tanh_f(float x) {
    return 2.0f / (1.0f + __expf(-2.0f * x)) - 1.0f;
}

__device__ __forceinline__ float dot4(const float4 w, const float4 h, float a) {
    a = __fmaf_rn(w.x, h.x, a);
    a = __fmaf_rn(w.y, h.y, a);
    a = __fmaf_rn(w.z, h.z, a);
    a = __fmaf_rn(w.w, h.w, a);
    return a;
}

// ---------------------------------------------------------------------------
// Phase 2: recurrent scan. Blocks [0,128) = forward dir, [128,256) = reverse.
// Each block owns 16 hidden units; each wave owns 2 (rows j, j+H, j+2H of
// w_hh -> 3 length-2048 dots). Lane owns contiguous float4 chunks:
// k = q*128 + lane*4  ->  3x global_load_b128 + 1x ds_load_b128 per iter,
// with global_prefetch_b8 issued ~1KB ahead on each weight stream.
// fused!=0: also computes the w_ih matvec inline (no precomputed igates).
// ---------------------------------------------------------------------------
__global__ __launch_bounds__(NT) void gru_scan(
    const float* __restrict__ seq,
    const float* __restrict__ wih_f, const float* __restrict__ b_f,
    const float* __restrict__ wih_r, const float* __restrict__ b_r,
    const float* __restrict__ whh_f, const float* __restrict__ bn_f,
    const float* __restrict__ whh_r, const float* __restrict__ bn_r,
    const float* __restrict__ igf, const float* __restrict__ igr,
    float* __restrict__ out, float* __restrict__ hbuf,
    unsigned* __restrict__ bar, int fused)
{
    __shared__ __align__(16) float hs[H_DIM]; // 8 KB: previous h (this dir)
    __shared__ __align__(16) float xs[Y_DIM]; // 2 KB: input row (fused path)

    const int lane = threadIdx.x & 31;
    const int wave = threadIdx.x >> 5;
    const int dir  = blockIdx.x >> 7;   // 0 fwd, 1 rev
    const int blk  = blockIdx.x & 127;

    const float* whh  = dir ? whh_r : whh_f;
    const float* bn   = dir ? bn_r  : bn_f;
    const float* wih  = dir ? wih_r : wih_f;
    const float* bias = dir ? b_r   : b_f;
    const float* ig   = dir ? igr   : igf;

    const int j0 = blk * 16 + wave * 2; // first of 2 hidden units for wave

    int cur = 0;
    for (int s = 0; s < T_LEN; ++s) {
        const float* hprev = hbuf + (size_t)(cur * 2 + dir) * H_DIM;
        float* hnext = hbuf + (size_t)((cur ^ 1) * 2 + dir) * H_DIM;

        // stage h_{s-1} (and x_s in fused mode) into LDS, 16B granules
        for (int i = threadIdx.x; i < H_DIM / 4; i += NT)
            ((float4*)hs)[i] = ((const float4*)hprev)[i];
        if (fused) {
            const size_t t = dir ? (size_t)(T_LEN - 1 - s) : (size_t)s;
            const float4* xrow = (const float4*)(seq + t * Y_DIM);
            for (int i = threadIdx.x; i < Y_DIM / 4; i += NT)
                ((float4*)xs)[i] = xrow[i];
        }
        __syncthreads();

#pragma unroll
        for (int u = 0; u < 2; ++u) {
            const int j = j0 + u;
            const float4* w0 = (const float4*)(whh + (size_t)j * H_DIM);
            const float4* w1 = (const float4*)(whh + (size_t)(j + H_DIM) * H_DIM);
            const float4* w2 = (const float4*)(whh + (size_t)(j + 2 * H_DIM) * H_DIM);
            const float4* hv = (const float4*)hs;

            float a0 = 0.f, a1 = 0.f, a2 = 0.f;
#pragma unroll 4
            for (int q = 0; q < H_DIM / 128; ++q) { // 16 iters
                const int kv = q * 32 + lane;       // float4 index
                // pull the next ~1KB of each weight stream toward the WGP
                __builtin_prefetch(w0 + kv + 64, 0, 0);
                __builtin_prefetch(w1 + kv + 64, 0, 0);
                __builtin_prefetch(w2 + kv + 64, 0, 0);
                const float4 h4 = hv[kv];
                a0 = dot4(w0[kv], h4, a0);
                a1 = dot4(w1[kv], h4, a1);
                a2 = dot4(w2[kv], h4, a2);
            }
            float g0 = 0.f, g1 = 0.f, g2 = 0.f;
            if (fused) {
                const float4* x0 = (const float4*)(wih + (size_t)j * Y_DIM);
                const float4* x1 = (const float4*)(wih + (size_t)(j + H_DIM) * Y_DIM);
                const float4* x2 = (const float4*)(wih + (size_t)(j + 2 * H_DIM) * Y_DIM);
                const float4* xv = (const float4*)xs;
#pragma unroll
                for (int q = 0; q < Y_DIM / 128; ++q) { // 4 iters
                    const int kv = q * 32 + lane;
                    const float4 x4 = xv[kv];
                    g0 = dot4(x0[kv], x4, g0);
                    g1 = dot4(x1[kv], x4, g1);
                    g2 = dot4(x2[kv], x4, g2);
                }
            }
#pragma unroll
            for (int off = 16; off; off >>= 1) {
                a0 += __shfl_xor(a0, off, 32);
                a1 += __shfl_xor(a1, off, 32);
                a2 += __shfl_xor(a2, off, 32);
            }
            if (fused) {
#pragma unroll
                for (int off = 16; off; off >>= 1) {
                    g0 += __shfl_xor(g0, off, 32);
                    g1 += __shfl_xor(g1, off, 32);
                    g2 += __shfl_xor(g2, off, 32);
                }
            }

            if (lane == 0) {
                float ir, iz, in_;
                if (fused) {
                    ir  = g0 + bias[j];
                    iz  = g1 + bias[j + H_DIM];
                    in_ = g2 + bias[j + 2 * H_DIM];
                } else {
                    const float* igrow = ig + (size_t)s * G3;
                    ir  = igrow[j];
                    iz  = igrow[j + H_DIM];
                    in_ = igrow[j + 2 * H_DIM];
                }
                const float r = sigm_f(ir + a0);
                const float z = sigm_f(iz + a1);
                const float n = tanh_f(in_ + r * (a2 + bn[j]));
                const float hp = hs[j];
                const float hnew = n + z * (hp - n);
                hnext[j] = hnew;
                const int t = dir ? (T_LEN - 1 - s) : s;
                out[(size_t)t * OUTW + dir * H_DIM + j] = hnew;
            }
        }
        grid_sync(bar, (unsigned)(s + 1));
        cur ^= 1;
    }
}

// ---------------------------------------------------------------------------
extern "C" void kernel_launch(void* const* d_in, const int* in_sizes, int n_in,
                              void* d_out, int out_size, void* d_ws,
                              size_t ws_size, hipStream_t stream)
{
    (void)in_sizes; (void)n_in; (void)out_size;
    const float* seq    = (const float*)d_in[0];
    const float* w_ih_f = (const float*)d_in[1];
    const float* w_hh_f = (const float*)d_in[2];
    const float* b_f    = (const float*)d_in[3];
    const float* bn_f   = (const float*)d_in[4];
    const float* w_ih_r = (const float*)d_in[5];
    const float* w_hh_r = (const float*)d_in[6];
    const float* b_r    = (const float*)d_in[7];
    const float* bn_r   = (const float*)d_in[8];
    float* out = (float*)d_out;

    const size_t igN  = (size_t)T_LEN * G3;
    const size_t need = 2 * igN * sizeof(float) + 4 * H_DIM * sizeof(float)
                        + 64 * sizeof(unsigned);
    const bool pre = (ws_size >= need);

    char* ws = (char*)d_ws;
    float* igf = nullptr;
    float* igr = nullptr;
    float* hbuf;
    if (pre) {
        igf  = (float*)ws;
        igr  = igf + igN;
        hbuf = igr + igN;
    } else {
        hbuf = (float*)ws;
    }
    unsigned* bar = (unsigned*)(hbuf + 4 * H_DIM);

    scan_init<<<dim3(32), dim3(256), 0, stream>>>(hbuf, bar);

    if (pre) {
        dim3 grid(G3 / 128, T_LEN / 64, 2);
        igates_gemm<<<grid, dim3(256), 0, stream>>>(
            seq, w_ih_f, b_f, w_ih_r, b_r, igf, igr);
    }

    gru_scan<<<dim3(NB), dim3(NT), 0, stream>>>(
        seq, w_ih_f, b_f, w_ih_r, b_r, w_hh_f, bn_f, w_hh_r, bn_r,
        igf, igr, out, hbuf, bar, pre ? 0 : 1);
}
